// MultiheadAttention_996432412848
// MI455X (gfx1250) — compile-verified, matrix operand order checked by reading, not run on hardware
//
#include <hip/hip_runtime.h>
#include <hip/hip_bf16.h>

typedef __attribute__((ext_vector_type(16))) _Float16 v16h;
typedef __attribute__((ext_vector_type(8)))  float    v8f;

#define E_DIM 1024
#define L_DIM 1024
#define N_B   4
#define H_N   16
#define HD    64
#define MROWS 4096   /* L*N */

static __device__ __forceinline__ v8f wmma_f16(v16h a, v16h b, v8f c) {
  // D = A(16x32 f16) * B(32x16 f16) + C(16x16 f32)
  return __builtin_amdgcn_wmma_f32_16x16x32_f16(false, a, false, b, (short)0, c,
                                                false, false);
}

// Load a 16x32 f16 A-fragment (or the symmetric B-fragment) from a row-major
// fp32 matrix, converting to f16. ISA layout: lane L holds row (L&15); K is
// striped as [kb..kb+7] and [kb+16..kb+23] with kb = (L>=16 ? 8 : 0).
static __device__ __forceinline__ v16h frag_f32(const float* __restrict__ base,
                                                int lane, int r0, int k0, int ld) {
  int r  = r0 + (lane & 15);
  int kb = k0 + ((lane >> 4) << 3);
  const float* p = base + (long)r * ld + kb;
  v16h a;
#pragma unroll
  for (int i = 0; i < 8; ++i) {
    a[i]     = (_Float16)p[i];
    a[i + 8] = (_Float16)p[i + 16];
  }
  return a;
}

static __device__ __forceinline__ v16h frag_f16(const _Float16* base,
                                                int lane, int r0, int k0, int ld) {
  int r  = r0 + (lane & 15);
  int kb = k0 + ((lane >> 4) << 3);
  const _Float16* p = base + (long)r * ld + kb;
  v16h a;
#pragma unroll
  for (int i = 0; i < 8; ++i) {
    a[i]     = p[i];
    a[i + 8] = p[i + 16];
  }
  return a;
}

// ---------------------------------------------------------------------------
__global__ __launch_bounds__(256) void zero_kernel(float* __restrict__ p, int n4) {
  int i = blockIdx.x * blockDim.x + threadIdx.x;
  if (i < n4) ((float4*)p)[i] = make_float4(0.f, 0.f, 0.f, 0.f);
}

// ---------------------------------------------------------------------------
// QKV projection: Y = X @ Wz^T + bz  (q additionally scaled by hd^-0.5)
// One wave computes a 32x32 tile. Outputs to per-head f16 layouts:
//   q_ws[b][l][d], k_ws[b][s][d], v_t[b][d][s]   (b = nb*16 + h)
__global__ __launch_bounds__(256) void qkv_proj_kernel(
    const float* __restrict__ q_in, const float* __restrict__ k_in,
    const float* __restrict__ v_in, const float* __restrict__ W,
    const float* __restrict__ bias,
    _Float16* __restrict__ q_ws, _Float16* __restrict__ k_ws,
    _Float16* __restrict__ v_t) {
  const int z = blockIdx.z;
  const float* X  = (z == 0) ? q_in : (z == 1) ? k_in : v_in;
  const float* Wz = W + (long)z * E_DIM * E_DIM;
  const float* bz = bias + z * E_DIM;
  const int lane = threadIdx.x & 31;
  const int wave = threadIdx.x >> 5;
  const int tile = blockIdx.x * 8 + wave;        // 4096 tiles (128 x 32)
  const int m0 = (tile >> 5) << 5;
  const int n0 = (tile & 31) << 5;

  v8f zacc = {};
  v8f c00 = zacc, c01 = zacc, c10 = zacc, c11 = zacc;
  for (int k0 = 0; k0 < E_DIM; k0 += 32) {
    if (k0 + 32 < E_DIM) {  // gfx1250 global_prefetch_b8 for next K slab
      __builtin_prefetch(X  + (long)(m0 + (lane & 15)) * E_DIM + k0 + 32, 0, 1);
      __builtin_prefetch(Wz + (long)(n0 + (lane & 15)) * E_DIM + k0 + 32, 0, 1);
    }
    v16h a0 = frag_f32(X,  lane, m0,      k0, E_DIM);
    v16h a1 = frag_f32(X,  lane, m0 + 16, k0, E_DIM);
    v16h b0 = frag_f32(Wz, lane, n0,      k0, E_DIM);
    v16h b1 = frag_f32(Wz, lane, n0 + 16, k0, E_DIM);
    c00 = wmma_f16(a0, b0, c00);
    c01 = wmma_f16(a0, b1, c01);
    c10 = wmma_f16(a1, b0, c10);
    c11 = wmma_f16(a1, b1, c11);
  }
  const float scale = (z == 0) ? 0.125f : 1.0f;
#pragma unroll
  for (int t = 0; t < 4; ++t) {
    v8f c = (t == 0) ? c00 : (t == 1) ? c01 : (t == 2) ? c10 : c11;
    int bm = m0 + ((t >> 1) << 4);
    int bn = n0 + ((t & 1) << 4);
#pragma unroll
    for (int r = 0; r < 8; ++r) {
      int m = bm + r + ((lane >> 4) << 3);     // C/D layout: VGPR r -> M=r or r+8
      int e = bn + (lane & 15);
      float val = (c[r] + bz[e]) * scale;
      int l = m >> 2, nb = m & 3, h = e >> 6, d = e & 63;
      int b = nb * H_N + h;
      if (z == 0)      q_ws[(long)b * (L_DIM * HD) + (long)l * HD + d]  = (_Float16)val;
      else if (z == 1) k_ws[(long)b * (L_DIM * HD) + (long)l * HD + d]  = (_Float16)val;
      else             v_t [(long)b * (L_DIM * HD) + (long)d * L_DIM + l] = (_Float16)val;
    }
  }
}

// ---------------------------------------------------------------------------
// Attention: one workgroup per (head-batch b, 16-row L block).
// scores(16x1024) -> LDS fp32 -> softmax -> P f16 in LDS (+ atomic head-avg
// into attn_weights) -> ctx(16x64) = P @ V with K split over 8 waves.
__global__ __launch_bounds__(256) void attn_kernel(
    const _Float16* __restrict__ q_ws, const _Float16* __restrict__ k_ws,
    const _Float16* __restrict__ v_t, _Float16* __restrict__ ctx_ws,
    float* __restrict__ attn_w) {
  extern __shared__ char smem[];
  float*    sc = (float*)smem;                       // 16*1024 fp32 (64 KB)
  _Float16* pf = (_Float16*)(smem + 16 * 1024 * 4);  // 16*1024 f16 (32 KB)
  __shared__ float rmax[16][16];
  __shared__ float rsum[16][16];
  __shared__ float rowmax[16];
  __shared__ float rowinv[16];

  const int Lb = blockIdx.x;       // 0..63 (16-row blocks of L)
  const int b  = blockIdx.y;       // 0..63 head-batches
  const int nb = b >> 4, h = b & 15;
  const int lane = threadIdx.x & 31;
  const int wave = threadIdx.x >> 5;

  const _Float16* qb = q_ws + (long)b * (L_DIM * HD) + (long)Lb * 16 * HD;
  const _Float16* kb = k_ws + (long)b * (L_DIM * HD);
  const _Float16* vb = v_t  + (long)b * (L_DIM * HD);

  // ---- Phase 1: scores (each wave: 16 x 128 slab) ----
  {
    v8f zacc = {};
    v8f acc[8];
#pragma unroll
    for (int j = 0; j < 8; ++j) acc[j] = zacc;
#pragma unroll
    for (int d0 = 0; d0 < HD; d0 += 32) {
      v16h a = frag_f16(qb, lane, 0, d0, HD);
#pragma unroll
      for (int j = 0; j < 8; ++j) {
        v16h bf = frag_f16(kb, lane, wave * 128 + j * 16, d0, HD);
        acc[j] = wmma_f16(a, bf, acc[j]);
      }
    }
#pragma unroll
    for (int j = 0; j < 8; ++j) {
      int s = wave * 128 + j * 16 + (lane & 15);
#pragma unroll
      for (int r = 0; r < 8; ++r) {
        int row = r + ((lane >> 4) << 3);
        sc[row * 1024 + s] = acc[j][r];
      }
    }
  }
  __syncthreads();

  // ---- Phase 2: softmax over S=1024 (16 threads per row, 64 cols each) ----
  {
    const int row = threadIdx.x >> 4;
    const int seg = threadIdx.x & 15;
    float* srow = sc + row * 1024 + seg * 64;
    float mx = -3.0e38f;
    for (int i = 0; i < 64; ++i) mx = fmaxf(mx, srow[i]);
    rmax[row][seg] = mx;
    __syncthreads();
    if (seg == 0) {
      float m = rmax[row][0];
      for (int j = 1; j < 16; ++j) m = fmaxf(m, rmax[row][j]);
      rowmax[row] = m;
    }
    __syncthreads();
    float m = rowmax[row];
    float sum = 0.f;
    for (int i = 0; i < 64; ++i) { float e = __expf(srow[i] - m); srow[i] = e; sum += e; }
    rsum[row][seg] = sum;
    __syncthreads();
    if (seg == 0) {
      float s = 0.f;
      for (int j = 0; j < 16; ++j) s += rsum[row][j];
      rowinv[row] = 1.0f / s;
    }
    __syncthreads();
    float inv = rowinv[row];
    _Float16* prow = pf + row * 1024 + seg * 64;
    float* awrow = attn_w + (long)nb * (L_DIM * L_DIM)
                 + (long)(Lb * 16 + row) * 1024 + seg * 64;
    for (int i = 0; i < 64; ++i) {
      float p = srow[i] * inv;
      prow[i] = (_Float16)p;
      unsafeAtomicAdd(&awrow[i], p * 0.0625f);   // head average (1/H)
    }
  }
  __syncthreads();

  // ---- Phase 3: ctx(16x64) = P(16x1024) @ V(1024x64), K split across waves ----
  {
    v8f zacc = {};
    v8f cc[4];
#pragma unroll
    for (int t = 0; t < 4; ++t) cc[t] = zacc;
    for (int ks = 0; ks < 4; ++ks) {
      int k0 = wave * 128 + ks * 32;
      v16h a = frag_f16(pf, lane, 0, k0, 1024);
#pragma unroll
      for (int dt = 0; dt < 4; ++dt) {
        v16h bf = frag_f16(vb, lane, dt * 16, k0, 1024);   // v_t rows are d
        cc[dt] = wmma_f16(a, bf, cc[dt]);
      }
    }
    // per-wave partials -> LDS (reuse score buffer): [wave][row][d]
#pragma unroll
    for (int dt = 0; dt < 4; ++dt) {
      int d = dt * 16 + (lane & 15);
#pragma unroll
      for (int r = 0; r < 8; ++r) {
        int row = r + ((lane >> 4) << 3);
        sc[wave * 1024 + row * 64 + d] = cc[dt][r];
      }
    }
  }
  __syncthreads();
  // sum the 8 wave partials, store ctx in (m=l*N+nb, e=h*64+d) f16
  for (int e = threadIdx.x; e < 1024; e += 256) {
    float s = 0.f;
#pragma unroll
    for (int w = 0; w < 8; ++w) s += sc[w * 1024 + e];
    int row = e >> 6, d = e & 63;
    int l = Lb * 16 + row;
    int m = l * N_B + nb;
    ctx_ws[(long)m * E_DIM + h * HD + d] = (_Float16)s;
  }
}

// ---------------------------------------------------------------------------
// Out projection: attn_output = ctx @ Wout^T + bout (fp32 stores to d_out)
__global__ __launch_bounds__(256) void out_proj_kernel(
    const _Float16* __restrict__ ctx_ws, const float* __restrict__ Wout,
    const float* __restrict__ bout, float* __restrict__ out) {
  const int lane = threadIdx.x & 31;
  const int wave = threadIdx.x >> 5;
  const int tile = blockIdx.x * 8 + wave;
  const int m0 = (tile >> 5) << 5;
  const int n0 = (tile & 31) << 5;
  v8f zacc = {};
  v8f c00 = zacc, c01 = zacc, c10 = zacc, c11 = zacc;
  for (int k0 = 0; k0 < E_DIM; k0 += 32) {
    v16h a0 = frag_f16(ctx_ws, lane, m0,      k0, E_DIM);
    v16h a1 = frag_f16(ctx_ws, lane, m0 + 16, k0, E_DIM);
    v16h b0 = frag_f32(Wout,   lane, n0,      k0, E_DIM);
    v16h b1 = frag_f32(Wout,   lane, n0 + 16, k0, E_DIM);
    c00 = wmma_f16(a0, b0, c00);
    c01 = wmma_f16(a0, b1, c01);
    c10 = wmma_f16(a1, b0, c10);
    c11 = wmma_f16(a1, b1, c11);
  }
#pragma unroll
  for (int t = 0; t < 4; ++t) {
    v8f c = (t == 0) ? c00 : (t == 1) ? c01 : (t == 2) ? c10 : c11;
    int bm = m0 + ((t >> 1) << 4);
    int bn = n0 + ((t & 1) << 4);
#pragma unroll
    for (int r = 0; r < 8; ++r) {
      int m = bm + r + ((lane >> 4) << 3);
      int e = bn + (lane & 15);
      out[(long)m * E_DIM + e] = c[r] + bout[e];
    }
  }
}

// ---------------------------------------------------------------------------
extern "C" void kernel_launch(void* const* d_in, const int* in_sizes, int n_in,
                              void* d_out, int out_size, void* d_ws, size_t ws_size,
                              hipStream_t stream) {
  const float* q_in = (const float*)d_in[0];
  const float* k_in = (const float*)d_in[1];
  const float* v_in = (const float*)d_in[2];
  const float* W    = (const float*)d_in[3];
  const float* bias = (const float*)d_in[4];
  const float* Wout = (const float*)d_in[5];
  const float* bout = (const float*)d_in[6];

  float* out    = (float*)d_out;          // attn_output: 4,194,304 floats
  float* attn_w = out + 4194304;          // attn_weights: 4,194,304 floats

  char* ws = (char*)d_ws;                 // 32 MB total
  _Float16* q_ws   = (_Float16*)(ws);                 //  8 MB
  _Float16* k_ws   = (_Float16*)(ws + (8u  << 20));   //  8 MB
  _Float16* v_t    = (_Float16*)(ws + (16u << 20));   //  8 MB
  _Float16* ctx_ws = (_Float16*)(ws + (24u << 20));   //  8 MB

  zero_kernel<<<4096, 256, 0, stream>>>(attn_w, 1048576);
  qkv_proj_kernel<<<dim3(512, 1, 3), 256, 0, stream>>>(q_in, k_in, v_in, W, bias,
                                                       q_ws, k_ws, v_t);
  attn_kernel<<<dim3(64, 64), 256, 96 * 1024, stream>>>(q_ws, k_ws, v_t, ctx_ws,
                                                        attn_w);
  out_proj_kernel<<<512, 256, 0, stream>>>(ctx_ws, Wout, bout, out);
}
